// InputFeedRNNDecoder_88802743812840
// MI455X (gfx1250) — matrix operand order repaired and checked by reading, not compiled, not run on hardware
//
#include <hip/hip_runtime.h>
#include <hip/hip_bf16.h>

typedef __bf16 bf16_t;
typedef __attribute__((ext_vector_type(16))) __bf16 v16bf;
typedef __attribute__((ext_vector_type(8)))  __bf16 v8bf;
typedef __attribute__((ext_vector_type(8)))  float  v8f;

#define B_ 64
#define T_ 64
#define S_ 256
#define E_ 1024
#define H_ 1024
#define NEGV (-1e9f)

__device__ __forceinline__ float sigm_(float x) { return 1.0f / (1.0f + __expf(-x)); }

// ---------------------------------------------------------------------------
// WMMA GEMM: C[M,N] = A[M,K] @ W[N,K]^T (+bias[N]); bf16 inputs, f32 accum.
// One wave per 16x16 tile; 4 waves/block along N; grid.z = batch.
// A-frag (16x32 bf16): lane half h (lane>>4), row=lane&15; per-lane K runs
//   [k0+8h .. +8) and [k0+8h+16 .. +8).  B-frag (32x16): col=lane&15; per-lane
//   K runs [k0+8h .. +8) and [k0+8h+16 .. +8).  C/D: vgpr r -> row r + 8h, col lane&15.
// ---------------------------------------------------------------------------
__global__ void k_gemm(const bf16_t* __restrict__ A, long long lda, long long sA,
                       const bf16_t* __restrict__ W, long long ldw, long long sW,
                       const float* __restrict__ bias,
                       void* __restrict__ Cp, long long ldc, long long sC, int cbf,
                       int M, int N, int K)
{
    int bz = blockIdx.z;
    A += (long long)bz * sA;
    W += (long long)bz * sW;
    int lane = threadIdx.x & 31;
    int wv   = threadIdx.x >> 5;
    int nt = blockIdx.x * (blockDim.x >> 5) + wv;
    int mt = blockIdx.y;
    if (nt * 16 >= N || mt * 16 >= M) return;   // wave-uniform: EXEC stays all-1s
    int half = lane >> 4, r = lane & 15;

    const bf16_t* arow = A + (long long)(mt * 16 + r) * lda;
    const bf16_t* wrow = W + (long long)(nt * 16 + r) * ldw;
    v8f acc = {};
    for (int k0 = 0; k0 < K; k0 += 32) {
        v8bf a0 = *(const v8bf*)(arow + k0 + half * 8);
        v8bf a1 = *(const v8bf*)(arow + k0 + half * 8 + 16);
        v8bf b0 = *(const v8bf*)(wrow + k0 + half * 8);
        v8bf b1 = *(const v8bf*)(wrow + k0 + half * 8 + 16);
        v16bf av, bv;
#pragma unroll
        for (int e = 0; e < 8; ++e) { av[e] = a0[e]; av[e + 8] = a1[e]; }
#pragma unroll
        for (int e = 0; e < 8; ++e) { bv[e] = b0[e]; bv[e + 8] = b1[e]; }
        acc = __builtin_amdgcn_wmma_f32_16x16x32_bf16(false, av, false, bv,
                                                      (short)0, acc, false, false);
    }
    float bb = bias ? bias[nt * 16 + r] : 0.0f;
    long long base = (long long)bz * sC + (long long)(mt * 16 + half * 8) * ldc + nt * 16 + r;
    if (cbf) {
        bf16_t* C = (bf16_t*)Cp;
#pragma unroll
        for (int rr = 0; rr < 8; ++rr) C[base + (long long)rr * ldc] = (bf16_t)(acc[rr] + bb);
    } else {
        float* C = (float*)Cp;
#pragma unroll
        for (int rr = 0; rr < 8; ++rr) C[base + (long long)rr * ldc] = acc[rr] + bb;
    }
}

// ------------------------- prep / utility kernels --------------------------
__global__ void k_f2b(bf16_t* d, const float* s, long long n)
{
    long long st = (long long)gridDim.x * blockDim.x;
    for (long long i = (long long)blockIdx.x * blockDim.x + threadIdx.x; i < n; i += st)
        d[i] = (bf16_t)s[i];
}

__global__ void k_transp(bf16_t* d, const float* s, int R, int C)
{   // d[C,R]: d[c*R+r] = s[r*C+c]
    long long n = (long long)R * C;
    long long st = (long long)gridDim.x * blockDim.x;
    for (long long i = (long long)blockIdx.x * blockDim.x + threadIdx.x; i < n; i += st) {
        int c = (int)(i / R), r = (int)(i % R);
        d[i] = (bf16_t)s[(long long)r * C + c];
    }
}

__global__ void k_pack2(bf16_t* d, const float* s0, int K0, const float* s1, int K1, int N)
{   // d[N, K0+K1] = [s0 | s1] rowwise, converted to bf16
    long long KT = K0 + K1;
    long long n = (long long)N * KT;
    long long st = (long long)gridDim.x * blockDim.x;
    for (long long i = (long long)blockIdx.x * blockDim.x + threadIdx.x; i < n; i += st) {
        long long nn = i / KT; int k = (int)(i % KT);
        float v = (k < K0) ? s0[nn * K0 + k] : s1[nn * K1 + (k - K0)];
        d[i] = (bf16_t)v;
    }
}

__global__ void k_addv(float* d, const float* a, const float* b, int n)
{
    for (int i = blockIdx.x * blockDim.x + threadIdx.x; i < n; i += gridDim.x * blockDim.x)
        d[i] = a[i] + b[i];
}

__global__ void k_copyf(float* d, const float* s, long long n)
{
    long long st = (long long)gridDim.x * blockDim.x;
    for (long long i = (long long)blockIdx.x * blockDim.x + threadIdx.x; i < n; i += st)
        d[i] = s[i];
}

__global__ void k_init(const float* h0, const float* c0,
                       float* h0s, float* h1s, float* c0s, float* c1s, float* fds,
                       bf16_t* xc0, bf16_t* xc1, bf16_t* xc2)
{
    int n = B_ * H_;
    for (int i = blockIdx.x * blockDim.x + threadIdx.x; i < n; i += gridDim.x * blockDim.x) {
        int b = i >> 10, h = i & 1023;
        float v0 = h0[i], v1 = h0[n + i];
        h0s[i] = v0; h1s[i] = v1;
        c0s[i] = c0[i]; c1s[i] = c0[n + i];
        fds[i] = 0.0f;
        xc0[b * 3072 + 1024 + h] = (bf16_t)0.0f;   // feed = 0
        xc0[b * 3072 + 2048 + h] = (bf16_t)v0;     // h_l0
        xc1[b * 2048 + h]        = (bf16_t)v0;     // h_l0
        xc1[b * 2048 + 1024 + h] = (bf16_t)v1;     // h_l1
        xc2[b * 2048 + 1024 + h] = (bf16_t)v1;     // out (= h_l1)
    }
}

__global__ void k_xt(bf16_t* xc0, const float* inputs, int t)
{
    int n = B_ * E_;
    for (int i = blockIdx.x * blockDim.x + threadIdx.x; i < n; i += gridDim.x * blockDim.x) {
        int b = i >> 10, e = i & 1023;
        xc0[b * 3072 + e] = (bf16_t)inputs[(long long)b * T_ * E_ + (long long)t * E_ + e];
    }
}

// ------------------------------ LSTM pointwise -----------------------------
__global__ void k_lstm(const float* __restrict__ g, float* __restrict__ c,
                       float* __restrict__ hst,
                       bf16_t* d0, long long ld0, bf16_t* d1, long long ld1,
                       bf16_t* d2, long long ld2, float* f0, long long lf0)
{
    int n = B_ * H_;
    for (int i = blockIdx.x * blockDim.x + threadIdx.x; i < n; i += gridDim.x * blockDim.x) {
        int b = i >> 10, h = i & 1023;
        const float* gr = g + (long long)b * 4096;
        float iv = sigm_(gr[h]);
        float fv = sigm_(gr[1024 + h]);
        float gv = tanhf(gr[2048 + h]);
        float ov = sigm_(gr[3072 + h]);
        float cn = fv * c[i] + iv * gv;
        c[i] = cn;
        float hn = ov * tanhf(cn);
        hst[i] = hn;
        bf16_t hb = (bf16_t)hn;
        if (d0) d0[(long long)b * ld0 + h] = hb;
        if (d1) d1[(long long)b * ld1 + h] = hb;
        if (d2) d2[(long long)b * ld2 + h] = hb;
        if (f0) f0[(long long)b * lf0 + h] = hn;
    }
}

// ----------------------------- std attention -------------------------------
__global__ void k_scores(const float* __restrict__ q, const float* __restrict__ mem,
                         const int* __restrict__ mask, float* __restrict__ sc)
{
    int b = blockIdx.x;
    int wv = threadIdx.x >> 5, lane = threadIdx.x & 31;
    int nw = blockDim.x >> 5;
    const float* qb = q + b * H_;
    const float* mb = mem + (long long)b * S_ * H_;
    for (int s = wv; s < S_; s += nw) {
        const float* row = mb + (long long)s * H_;
        float acc = 0.0f;
        for (int h = lane; h < H_; h += 32) acc += qb[h] * row[h];
#pragma unroll
        for (int off = 16; off > 0; off >>= 1) acc += __shfl_xor(acc, off, 32);
        if (lane == 0) sc[b * S_ + s] = (mask[b * S_ + s] > 0) ? acc : NEGV;
    }
}

__global__ void k_softmax_s(const float* sc, float* a, float* o_std, int t)
{
    __shared__ float red[S_];
    int b = blockIdx.x, tid = threadIdx.x;
    float v = sc[b * S_ + tid];
    red[tid] = v; __syncthreads();
    for (int o = 128; o > 0; o >>= 1) { if (tid < o) red[tid] = fmaxf(red[tid], red[tid + o]); __syncthreads(); }
    float m = red[0]; __syncthreads();
    float e = __expf(v - m);
    red[tid] = e; __syncthreads();
    for (int o = 128; o > 0; o >>= 1) { if (tid < o) red[tid] += red[tid + o]; __syncthreads(); }
    float p = e / red[0];
    a[b * S_ + tid] = p;
    o_std[(long long)b * T_ * S_ + (long long)t * S_ + tid] = p;
}

__global__ void k_ctx(const float* __restrict__ a, const float* __restrict__ mem,
                      bf16_t* __restrict__ xc2)
{
    __shared__ float sa[S_];
    int b = blockIdx.x, tid = threadIdx.x;
    sa[tid] = a[b * S_ + tid];
    __syncthreads();
    const float* mb = mem + (long long)b * S_ * H_;
    for (int h = tid; h < H_; h += blockDim.x) {
        float acc = 0.0f;
        for (int s = 0; s < S_; ++s) acc += sa[s] * mb[(long long)s * H_ + h];
        xc2[b * 2048 + h] = (bf16_t)acc;
    }
}

__global__ void k_tanhfeed(const float* __restrict__ pre, float* o_outs, bf16_t* outb,
                           bf16_t* xc0, float* fds, int t)
{
    int n = B_ * H_;
    for (int i = blockIdx.x * blockDim.x + threadIdx.x; i < n; i += gridDim.x * blockDim.x) {
        int b = i >> 10, h = i & 1023;
        float v = tanhf(pre[i]);
        o_outs[(long long)b * T_ * H_ + (long long)t * H_ + h] = v;
        outb[(long long)b * T_ * H_ + (long long)t * H_ + h] = (bf16_t)v;
        xc0[b * 3072 + 1024 + h] = (bf16_t)v;   // input feed for next step
        fds[i] = v;
    }
}

// --------------------------- copy / coref epilogues ------------------------
__global__ void k_copy_softmax(const bf16_t* __restrict__ P, const float* __restrict__ cu,
                               const float* __restrict__ csc, const int* __restrict__ mask,
                               float* __restrict__ out)
{
    __shared__ float red[256];
    __shared__ float scv[257];
    int bt = blockIdx.x;
    int b = bt >> 6, t = bt & 63;
    int tid = threadIdx.x;
    // col 0: dot(P[b,t,:], copy_unknown)
    const bf16_t* pr = P + (long long)bt * H_;
    float acc = 0.0f;
    for (int h = tid; h < H_; h += 256) acc += (float)pr[h] * cu[h];
    red[tid] = acc; __syncthreads();
    for (int o = 128; o > 0; o >>= 1) { if (tid < o) red[tid] += red[tid + o]; __syncthreads(); }
    if (tid == 0) scv[0] = red[0];
    float v = csc[((long long)b * T_ + t) * S_ + tid];
    scv[tid + 1] = (mask[b * S_ + tid] > 0) ? v : NEGV;
    __syncthreads();
    float own = scv[tid];
    float extra = (tid == 0) ? scv[256] : NEGV;
    red[tid] = fmaxf(own, extra); __syncthreads();
    for (int o = 128; o > 0; o >>= 1) { if (tid < o) red[tid] = fmaxf(red[tid], red[tid + o]); __syncthreads(); }
    float m = red[0]; __syncthreads();
    float e0 = __expf(own - m);
    float e256 = (tid == 0) ? __expf(scv[256] - m) : 0.0f;
    red[tid] = e0 + e256; __syncthreads();
    for (int o = 128; o > 0; o >>= 1) { if (tid < o) red[tid] += red[tid + o]; __syncthreads(); }
    float inv = 1.0f / red[0];
    float* dst = out + (long long)bt * 257;
    dst[tid] = e0 * inv;
    if (tid == 0) dst[256] = e256 * inv;
}

__global__ void k_coref_softmax(const bf16_t* __restrict__ R, const float* __restrict__ cna,
                                const float* __restrict__ ksc, float* __restrict__ out)
{
    __shared__ float red[256];
    int bt = blockIdx.x;
    int b = bt >> 6, t = bt & 63;
    int tid = threadIdx.x;
    const bf16_t* rr = R + (long long)bt * H_;
    float acc = 0.0f;
    for (int h = tid; h < H_; h += 256) acc += (float)rr[h] * cna[h];
    red[tid] = acc; __syncthreads();
    for (int o = 128; o > 0; o >>= 1) { if (tid < o) red[tid] += red[tid + o]; __syncthreads(); }
    if (tid == 0) {
        float s0 = red[0];
        const float* kr = ksc + ((long long)b * T_ + t) * T_;
        float m = s0;
        for (int j = 1; j <= t; ++j) m = fmaxf(m, kr[j - 1]);
        float sum = __expf(s0 - m);
        for (int j = 1; j <= t; ++j) sum += __expf(kr[j - 1] - m);
        float inv = 1.0f / sum;
        float* dst = out + (long long)bt * (T_ + 1);
        dst[0] = __expf(s0 - m) * inv;
        for (int j = 1; j <= T_; ++j)
            dst[j] = (j <= t) ? __expf(kr[j - 1] - m) * inv : 0.0f;
    }
}

// ---------------------------------------------------------------------------
extern "C" void kernel_launch(void* const* d_in, const int* in_sizes, int n_in,
                              void* d_out_, int out_size, void* d_ws, size_t ws_size,
                              hipStream_t stream)
{
    const float* inputs  = (const float*)d_in[0];
    const float* memf    = (const float*)d_in[1];
    const int*   mask    = (const int*)d_in[2];
    const float* h0      = (const float*)d_in[3];
    const float* c0      = (const float*)d_in[4];
    const float* cu      = (const float*)d_in[5];
    const float* cna     = (const float*)d_in[6];
    const float* wih0    = (const float*)d_in[7];
    const float* whh0    = (const float*)d_in[8];
    const float* bih0    = (const float*)d_in[9];
    const float* bhh0    = (const float*)d_in[10];
    const float* wih1    = (const float*)d_in[11];
    const float* whh1    = (const float*)d_in[12];
    const float* bih1    = (const float*)d_in[13];
    const float* bhh1    = (const float*)d_in[14];
    const float* attnWa  = (const float*)d_in[15];
    const float* attnWo  = (const float*)d_in[16];
    const float* copyWa  = (const float*)d_in[17];
    const float* corefWa = (const float*)d_in[18];
    float* out = (float*)d_out_;

    // output regions (return-order concat)
    float* o_outs  = out;                 // B*T*H
    float* o_rnn   = out + 4194304;       // B*T*H
    float* o_coref = out + 8388608;       // B*T*65
    float* o_copy  = out + 8654848;       // B*T*257
    float* o_std   = out + 9707520;       // B*T*S
    float* o_hn    = out + 10756096;      // 2*B*H
    float* o_cn    = out + 10887168;      // 2*B*H
    float* o_feed  = out + 11018240;      // B*1*H

    // workspace carve (~125 MB)
    char* ws = (char*)d_ws;
    size_t cur = 0;
    auto carve = [&](size_t bytes) -> char* {
        char* p = ws + cur;
        cur += (bytes + 255) & ~(size_t)255;
        return p;
    };
    bf16_t* wg0  = (bf16_t*)carve(2ull * 4096 * 3072);   // [wih0|whh0]
    bf16_t* wg1  = (bf16_t*)carve(2ull * 4096 * 2048);   // [wih1|whh1]
    bf16_t* wo   = (bf16_t*)carve(2ull * 1024 * 2048);   // attn_Wo
    bf16_t* waT  = (bf16_t*)carve(2ull * 1024 * 1024);   // attn_Wa^T
    bf16_t* cwaT = (bf16_t*)carve(2ull * 1024 * 1024);   // copy_Wa^T
    bf16_t* kwaT = (bf16_t*)carve(2ull * 1024 * 1024);   // coref_Wa^T
    bf16_t* memb = (bf16_t*)carve(2ull * B_ * S_ * H_);  // memory_bank bf16
    bf16_t* rnnb = (bf16_t*)carve(2ull * B_ * T_ * H_);  // rnn_outs bf16
    bf16_t* outb = (bf16_t*)carve(2ull * B_ * T_ * H_);  // outs bf16
    bf16_t* xc0  = (bf16_t*)carve(2ull * B_ * 3072);     // [x_t|feed|h0]
    bf16_t* xc1  = (bf16_t*)carve(2ull * B_ * 2048);     // [h0|h1]
    bf16_t* xc2  = (bf16_t*)carve(2ull * B_ * 2048);     // [ctx|out]
    bf16_t* pbf  = (bf16_t*)carve(2ull * B_ * T_ * H_);  // P / R projections
    float* bg0  = (float*)carve(4ull * 4096);
    float* bg1  = (float*)carve(4ull * 4096);
    float* g0   = (float*)carve(4ull * B_ * 4096);
    float* g1   = (float*)carve(4ull * B_ * 4096);
    float* qb   = (float*)carve(4ull * B_ * H_);
    float* scb  = (float*)carve(4ull * B_ * S_);
    float* ab   = (float*)carve(4ull * B_ * S_);
    float* pre  = (float*)carve(4ull * B_ * H_);
    float* c0s  = (float*)carve(4ull * B_ * H_);
    float* c1s  = (float*)carve(4ull * B_ * H_);
    float* h0s  = (float*)carve(4ull * B_ * H_);
    float* h1s  = (float*)carve(4ull * B_ * H_);
    float* fds  = (float*)carve(4ull * B_ * H_);
    float* cscb = (float*)carve(4ull * B_ * T_ * S_);
    float* kscb = (float*)carve(4ull * B_ * T_ * T_);
    (void)ws_size; (void)in_sizes; (void)n_in; (void)out_size;

    dim3 blk(256);
    auto grid1 = [](long long n) {
        long long g = (n + 255) / 256;
        if (g > 32768) g = 32768;
        return dim3((unsigned)g);
    };
    auto gemm = [&](const bf16_t* A, long long lda, long long sA,
                    const bf16_t* W, long long ldw, long long sW,
                    const float* bias, void* C, long long ldc, long long sC, int cbf,
                    int M, int N, int K, int batch) {
        dim3 g((unsigned)((N / 16 + 3) / 4), (unsigned)(M / 16), (unsigned)batch);
        k_gemm<<<g, dim3(128), 0, stream>>>(A, lda, sA, W, ldw, sW, bias, C, ldc, sC, cbf, M, N, K);
    };

    // -------- one-time prep (deterministic; runs every call) --------
    k_pack2<<<grid1(4096ll * 3072), blk, 0, stream>>>(wg0, wih0, 2048, whh0, 1024, 4096);
    k_pack2<<<grid1(4096ll * 2048), blk, 0, stream>>>(wg1, wih1, 1024, whh1, 1024, 4096);
    k_f2b  <<<grid1(1024ll * 2048), blk, 0, stream>>>(wo, attnWo, 1024ll * 2048);
    k_transp<<<grid1(1024ll * 1024), blk, 0, stream>>>(waT,  attnWa,  1024, 1024);
    k_transp<<<grid1(1024ll * 1024), blk, 0, stream>>>(cwaT, copyWa,  1024, 1024);
    k_transp<<<grid1(1024ll * 1024), blk, 0, stream>>>(kwaT, corefWa, 1024, 1024);
    k_f2b  <<<grid1((long long)B_ * S_ * H_), blk, 0, stream>>>(memb, memf, (long long)B_ * S_ * H_);
    k_addv <<<grid1(4096), blk, 0, stream>>>(bg0, bih0, bhh0, 4096);
    k_addv <<<grid1(4096), blk, 0, stream>>>(bg1, bih1, bhh1, 4096);
    k_init <<<grid1(B_ * H_), blk, 0, stream>>>(h0, c0, h0s, h1s, c0s, c1s, fds, xc0, xc1, xc2);

    // -------- sequential scan over T --------
    for (int t = 0; t < T_; ++t) {
        k_xt<<<grid1(B_ * E_), blk, 0, stream>>>(xc0, inputs, t);
        // layer-0 gates: [64,3072] @ [4096,3072]^T + bias
        gemm(xc0, 3072, 0, wg0, 3072, 0, bg0, g0, 4096, 0, 0, 64, 4096, 3072, 1);
        k_lstm<<<grid1(B_ * H_), blk, 0, stream>>>(g0, c0s, h0s,
              xc0 + 2048, 3072ll, xc1, 2048ll, (bf16_t*)nullptr, 0ll,
              (float*)nullptr, 0ll);
        // layer-1 gates: [64,2048] @ [4096,2048]^T + bias
        gemm(xc1, 2048, 0, wg1, 2048, 0, bg1, g1, 4096, 0, 0, 64, 4096, 2048, 1);
        k_lstm<<<grid1(B_ * H_), blk, 0, stream>>>(g1, c1s, h1s,
              xc1 + 1024, 2048ll, xc2 + 1024, 2048ll,
              rnnb + (long long)t * H_, (long long)T_ * H_,
              o_rnn + (long long)t * H_, (long long)T_ * H_);
        // q = out @ attn_Wa : [64,1024] @ [1024,1024]^T(WaT)
        gemm(xc1 + 1024, 2048, 0, waT, 1024, 0, nullptr, qb, 1024, 0, 0, 64, 1024, 1024, 1);
        k_scores   <<<dim3(B_), dim3(256), 0, stream>>>(qb, memf, mask, scb);
        k_softmax_s<<<dim3(B_), dim3(256), 0, stream>>>(scb, ab, o_std, t);
        k_ctx      <<<dim3(B_), dim3(256), 0, stream>>>(ab, memf, xc2);
        // outp = tanh([ctx|out] @ attn_Wo^T)
        gemm(xc2, 2048, 0, wo, 2048, 0, nullptr, pre, 1024, 0, 0, 64, 1024, 2048, 1);
        k_tanhfeed<<<grid1(B_ * H_), blk, 0, stream>>>(pre, o_outs, outb, xc0, fds, t);
    }

    // -------- copy attention (batch-parallel) --------
    // P = outs @ copy_Wa  -> bf16 [B*T, H]
    gemm(outb, 1024, 0, cwaT, 1024, 0, nullptr, pbf, 1024, 0, 1, B_ * T_, 1024, 1024, 1);
    // csc[b] = P[b] @ mem[b]^T : batched [64,1024]x[256,1024]^T
    gemm(pbf, 1024, (long long)T_ * H_, memb, 1024, (long long)S_ * H_,
         nullptr, cscb, S_, (long long)T_ * S_, 0, T_, S_, 1024, B_);
    k_copy_softmax<<<dim3(B_ * T_), dim3(256), 0, stream>>>(pbf, cu, cscb, mask, o_copy);

    // -------- coref attention (batch-parallel) --------
    // R = rnn_outs @ coref_Wa -> bf16 (reuse pbf)
    gemm(rnnb, 1024, 0, kwaT, 1024, 0, nullptr, pbf, 1024, 0, 1, B_ * T_, 1024, 1024, 1);
    // ksc[b] = R[b] @ rnn_outs[b]^T : batched [64,1024]x[64,1024]^T
    gemm(pbf, 1024, (long long)T_ * H_, rnnb, 1024, (long long)T_ * H_,
         nullptr, kscb, T_, (long long)T_ * T_, 0, T_, T_, 1024, B_);
    k_coref_softmax<<<dim3(B_ * T_), dim3(256), 0, stream>>>(pbf, cna, kscb, o_coref);

    // -------- final states --------
    k_copyf<<<grid1(B_ * H_), blk, 0, stream>>>(o_hn,            h0s, (long long)B_ * H_);
    k_copyf<<<grid1(B_ * H_), blk, 0, stream>>>(o_hn + B_ * H_,  h1s, (long long)B_ * H_);
    k_copyf<<<grid1(B_ * H_), blk, 0, stream>>>(o_cn,            c0s, (long long)B_ * H_);
    k_copyf<<<grid1(B_ * H_), blk, 0, stream>>>(o_cn + B_ * H_,  c1s, (long long)B_ * H_);
    k_copyf<<<grid1(B_ * H_), blk, 0, stream>>>(o_feed,          fds, (long long)B_ * H_);
}
